// MyModel_63823214019176
// MI455X (gfx1250) — compile-verified, matrix operand order checked by reading, not run on hardware
//
#include <hip/hip_runtime.h>
#include <hip/hip_bf16.h>

// ---------------------------------------------------------------------------
// LSTM (Keras gate order i,f,g,o) over SEQ=59 steps, B=4096, FEAT=64, U=1024,
// then dense->softmax to VOCAB=578. bf16 WMMA (v_wmma_f32_16x16x32_bf16),
// f32 accumulation + f32 cell state. Register double-buffered K pipeline.
// ---------------------------------------------------------------------------

#define BATCH   4096
#define SEQ     59
#define FEAT    64
#define UNITS   1024
#define VOCAB   578
#define VT      37            // ceil(578/16) column tiles (592 padded cols)
#define MT      (BATCH / 16)  // 256 row tiles
#define GATES4U (4 * UNITS)   // 4096

typedef __attribute__((ext_vector_type(16))) __bf16 v16bf;
typedef __attribute__((ext_vector_type(8)))  float  v8f;

__device__ __forceinline__ unsigned short f2bf(float f) {
    unsigned u = __float_as_uint(f);
    unsigned r = u + 0x7FFFu + ((u >> 16) & 1u);   // round-to-nearest-even
    return (unsigned short)(r >> 16);
}
__device__ __forceinline__ float sigm_f(float x) { return 1.f / (1.f + __expf(-x)); }
__device__ __forceinline__ float tanh_f(float x) { return 2.f / (1.f + __expf(-2.f * x)) - 1.f; }

// RA = reuse-A OPSEL hint: legal when consecutive WMMAs are the identical
// opcode and share the A operand (ISA 7.12 VOP3P fields).
template <bool RA>
__device__ __forceinline__ v8f wmma_bf16(v16bf a, v16bf b, v8f c) {
    return __builtin_amdgcn_wmma_f32_16x16x32_bf16(false, a, false, b,
                                                   (short)0, c, RA, false);
}

// --- Fragment layout helpers (CDNA5 ISA 16-bit A 16x32 / B 32x16 tables) ----
// Storage: one tile = 512 contiguous bf16; lane L owns elements [L*16, L*16+16).
// slot s within a lane: vgpr = s/2, halfword = s%2.
// k_local(lane,slot) = (vgpr&4 ? 16:0) + (vgpr&3)*2 + (lane>=16 ? 8:0) + halfword

// Convert row-major f32 weight W[K][ldw] into B-fragment tiles: out[kc][nt][512]
__global__ void conv_w_tiles(const float* __restrict__ W, int K, int N, int ldw,
                             int Ntiles, int total, unsigned short* __restrict__ out) {
    int idx = blockIdx.x * blockDim.x + threadIdx.x;
    if (idx >= total) return;
    int slot = idx & 15;
    int lane = (idx >> 4) & 31;
    int tile = idx >> 9;
    int nt = tile % Ntiles;
    int kc = tile / Ntiles;
    int nl = lane & 15, half = lane >> 4;
    int vg = slot >> 1, pos = slot & 1;
    int kl = ((vg & 4) ? 16 : 0) + (vg & 3) * 2 + half * 8 + pos;
    int k = kc * 32 + kl, n = nt * 16 + nl;
    float v = (k < K && n < N) ? W[(size_t)k * ldw + n] : 0.f;
    out[idx] = f2bf(v);
}

// Convert x f32 [B][SEQ][FEAT] into A-fragment tiles: xa[t][mtile][kc(2)][512]
__global__ void conv_x_tiles(const float* __restrict__ x,
                             unsigned short* __restrict__ xa, int total) {
    int idx = blockIdx.x * blockDim.x + threadIdx.x;
    if (idx >= total) return;
    int slot = idx & 15;
    int lane = (idx >> 4) & 31;
    int tile = idx >> 9;             // tile = (t*MT + mtile)*2 + kc
    int kc = tile & 1;
    int mt = (tile >> 1) & (MT - 1); // MT == 256
    int t  = tile >> 9;
    int ml = lane & 15, half = lane >> 4;
    int vg = slot >> 1, pos = slot & 1;
    int kl = ((vg & 4) ? 16 : 0) + (vg & 3) * 2 + half * 8 + pos;
    int k = kc * 32 + kl;            // 0..63
    int row = mt * 16 + ml;
    xa[idx] = f2bf(x[((size_t)row * SEQ + t) * FEAT + k]);
}

// ---------------------------------------------------------------------------
// One LSTM timestep: block = 256 thr (8 waves), computes 64 rows x 32 units,
// all 4 gates, fused cell/hidden update. Grid (BATCH/64, UNITS/32).
// K-loop is register double-buffered: loads for chunk k+1 in flight while the
// WMMA quartet for chunk k executes (partial s_wait_loadcnt, no full drains).
// ---------------------------------------------------------------------------
__global__ void __launch_bounds__(256)
lstm_step_kernel(const unsigned short* __restrict__ xa_t,   // [MT][2][512]
                 const unsigned short* __restrict__ wk,     // [2][256][512]
                 const unsigned short* __restrict__ ha_prev,// [MT][32][512] (+slack)
                 const unsigned short* __restrict__ wr,     // [32][256][512] (+slack)
                 const float* __restrict__ bias,            // [4096]
                 float* __restrict__ c,                     // [B][U]
                 unsigned short* __restrict__ ha_next) {    // [MT][32][512]
    const int tid  = threadIdx.x;
    const int wave = tid >> 5;
    const int lane = tid & 31;
    const int m0 = blockIdx.x * 64;
    const int n0 = blockIdx.y * 32;
    const int rb = wave >> 1, cb = wave & 1;
    const int mtile = (m0 >> 4) + rb;
    const int nu0 = n0 + cb * 16;
    const int ntu = nu0 >> 4;            // unit column tile 0..63
    const int loff = lane * 16;

    v8f acc0 = {}, acc1 = {}, acc2 = {}, acc3 = {};

    // ---- x_t @ kernel : K = 64 (2 chunks) ----
#pragma unroll
    for (int kc = 0; kc < 2; ++kc) {
        v16bf a = *(const v16bf*)(xa_t + (((mtile << 1) + kc) << 9) + loff);
        const unsigned short* wb = wk + ((size_t)(kc * 256) << 9) + loff;
        v16bf b0 = *(const v16bf*)(wb + ((size_t)(0 * 64 + ntu) << 9));
        v16bf b1 = *(const v16bf*)(wb + ((size_t)(1 * 64 + ntu) << 9));
        v16bf b2v = *(const v16bf*)(wb + ((size_t)(2 * 64 + ntu) << 9));
        v16bf b3 = *(const v16bf*)(wb + ((size_t)(3 * 64 + ntu) << 9));
        acc0 = wmma_bf16<true >(a, b0, acc0);
        acc1 = wmma_bf16<true >(a, b1, acc1);
        acc2 = wmma_bf16<true >(a, b2v, acc2);
        acc3 = wmma_bf16<false>(a, b3, acc3);
    }

    // ---- h @ rec_kernel : K = 1024 (32 chunks), ping-pong pipelined ----
    const unsigned short* ap  = ha_prev + (((size_t)mtile << 5) << 9) + loff;
    const unsigned short* w0p = wr + (((size_t)(0 * 64 + ntu)) << 9) + loff;
    const unsigned short* w1p = wr + (((size_t)(1 * 64 + ntu)) << 9) + loff;
    const unsigned short* w2p = wr + (((size_t)(2 * 64 + ntu)) << 9) + loff;
    const unsigned short* w3p = wr + (((size_t)(3 * 64 + ntu)) << 9) + loff;
    // chunk strides (elements): A tile = 1<<9, weight chunk = 256 tiles = 1<<17
    v16bf aA  = *(const v16bf*)(ap);
    v16bf b0A = *(const v16bf*)(w0p);
    v16bf b1A = *(const v16bf*)(w1p);
    v16bf b2A = *(const v16bf*)(w2p);
    v16bf b3A = *(const v16bf*)(w3p);
#pragma unroll 1
    for (int kc = 0; kc < 32; kc += 2) {
        const size_t o1 = (size_t)(kc + 1) << 17, o1a = (size_t)(kc + 1) << 9;
        const size_t o2 = (size_t)(kc + 2) << 17, o2a = (size_t)(kc + 2) << 9;
        // issue loads for chunk kc+1 (set B) before consuming set A
        v16bf aB  = *(const v16bf*)(ap + o1a);
        v16bf b0B = *(const v16bf*)(w0p + o1);
        v16bf b1B = *(const v16bf*)(w1p + o1);
        v16bf b2B = *(const v16bf*)(w2p + o1);
        v16bf b3B = *(const v16bf*)(w3p + o1);
        acc0 = wmma_bf16<true >(aA, b0A, acc0);
        acc1 = wmma_bf16<true >(aA, b1A, acc1);
        acc2 = wmma_bf16<true >(aA, b2A, acc2);
        acc3 = wmma_bf16<false>(aA, b3A, acc3);
        // issue loads for chunk kc+2 (set A); kc+2==32 reads slack chunk (unused)
        aA  = *(const v16bf*)(ap + o2a);
        b0A = *(const v16bf*)(w0p + o2);
        b1A = *(const v16bf*)(w1p + o2);
        b2A = *(const v16bf*)(w2p + o2);
        b3A = *(const v16bf*)(w3p + o2);
        acc0 = wmma_bf16<true >(aB, b0B, acc0);
        acc1 = wmma_bf16<true >(aB, b1B, acc1);
        acc2 = wmma_bf16<true >(aB, b2B, acc2);
        acc3 = wmma_bf16<false>(aB, b3B, acc3);
    }

    // ---- gates + cell update. C/D layout: m = v + 8*(lane/16), n = lane%16 ----
    const int nl = lane & 15, half = lane >> 4;
    const int nu = nu0 + nl;
    const float bi = bias[nu];
    const float bf = bias[UNITS + nu];
    const float bg = bias[2 * UNITS + nu];
    const float bo = bias[3 * UNITS + nu];
    const int mbase = m0 + rb * 16 + half * 8;
#pragma unroll
    for (int v = 0; v < 8; ++v) {
        int m = mbase + v;
        float ig = sigm_f(acc0[v] + bi);
        float fg = sigm_f(acc1[v] + bf);
        float gg = tanh_f(acc2[v] + bg);
        float og = sigm_f(acc3[v] + bo);
        size_t cidx = (size_t)m * UNITS + nu;
        float cn = fg * c[cidx] + ig * gg;
        c[cidx] = cn;
        float hn = og * tanh_f(cn);
        // scatter into next step's A-fragment layout
        int kk = nu & 31;
        int vg = ((kk >> 4) << 2) + ((kk & 7) >> 1);
        int lw = ((kk >> 3) & 1) * 16 + (m & 15);
        int slot = vg * 2 + (kk & 1);
        ha_next[((((size_t)(m >> 4) << 5) + (nu >> 5)) << 9) + lw * 16 + slot] = f2bf(hn);
    }
}

// ---------------------------------------------------------------------------
// Final dense (h_last @ w2 + b2) + row softmax. Block = 16 rows, grid = MT.
// ---------------------------------------------------------------------------
__global__ void __launch_bounds__(256)
dense_softmax_kernel(const unsigned short* __restrict__ ha,   // [MT][32][512] (+slack)
                     const unsigned short* __restrict__ w2t,  // [32][VT][512] (+slack)
                     const float* __restrict__ b2,
                     float* __restrict__ out) {               // [B][VOCAB]
    __shared__ float sl[16][600];
    const int tid = threadIdx.x, wave = tid >> 5, lane = tid & 31;
    const int mtile = blockIdx.x;
    const int loff = lane * 16;

    for (int ct = wave; ct < VT; ct += 8) {
        const unsigned short* hap = ha + (((size_t)mtile << 5) << 9) + loff;
        const unsigned short* wp  = w2t + ((size_t)ct << 9) + loff;
        v8f acc = {};
        v16bf a0 = *(const v16bf*)(hap);
        v16bf b0 = *(const v16bf*)(wp);
#pragma unroll 1
        for (int kc = 0; kc < 32; kc += 2) {
            v16bf a1 = *(const v16bf*)(hap + ((size_t)(kc + 1) << 9));
            v16bf b1 = *(const v16bf*)(wp + (((size_t)(kc + 1) * VT) << 9));
            acc = wmma_bf16<false>(a0, b0, acc);
            a0 = *(const v16bf*)(hap + ((size_t)(kc + 2) << 9));        // kc+2==32 -> slack
            b0 = *(const v16bf*)(wp + (((size_t)(kc + 2) * VT) << 9));  // kc+2==32 -> slack
            acc = wmma_bf16<false>(a1, b1, acc);
        }
        int nl = lane & 15, half = lane >> 4;
        int col = ct * 16 + nl;
        float badd = (col < VOCAB) ? b2[col] : -1e30f;
#pragma unroll
        for (int v = 0; v < 8; ++v) sl[v + half * 8][col] = acc[v] + badd;
    }
    __syncthreads();

#pragma unroll
    for (int rr = 0; rr < 2; ++rr) {
        int r = wave * 2 + rr;
        float mx = -1e30f;
        for (int cx = lane; cx < VOCAB; cx += 32) mx = fmaxf(mx, sl[r][cx]);
        for (int off = 16; off; off >>= 1) mx = fmaxf(mx, __shfl_xor(mx, off, 32));
        float s = 0.f;
        for (int cx = lane; cx < VOCAB; cx += 32) s += __expf(sl[r][cx] - mx);
        for (int off = 16; off; off >>= 1) s += __shfl_xor(s, off, 32);
        float inv = 1.f / s;
        size_t row = (size_t)mtile * 16 + r;
        for (int cx = lane; cx < VOCAB; cx += 32)
            out[row * VOCAB + cx] = __expf(sl[r][cx] - mx) * inv;
    }
}

// ---------------------------------------------------------------------------
extern "C" void kernel_launch(void* const* d_in, const int* in_sizes, int n_in,
                              void* d_out, int out_size, void* d_ws, size_t ws_size,
                              hipStream_t stream) {
    const float* x      = (const float*)d_in[0];  // [B, SEQ, FEAT]
    const float* kernel = (const float*)d_in[1];  // [FEAT, 4U]
    const float* rec    = (const float*)d_in[2];  // [U, 4U]
    const float* bias   = (const float*)d_in[3];  // [4U]
    const float* w2     = (const float*)d_in[4];  // [U, VOCAB]
    const float* b2     = (const float*)d_in[5];  // [VOCAB]
    float* out = (float*)d_out;

    // carve workspace (256B aligned)
    size_t off = 0;
    char* base = (char*)d_ws;
    auto take = [&](size_t bytes) -> char* {
        char* p = base + off;
        off += (bytes + 255) & ~(size_t)255;
        return p;
    };
    const size_t WK_ELEMS   = (size_t)2  * 256 * 512;          // 262144
    const size_t WR_ELEMS   = (size_t)32 * 256 * 512;          // 4194304
    const size_t WR_ALLOC   = WR_ELEMS + (size_t)256 * 512;    // +1 chunk slack
    const size_t W2T_ELEMS  = (size_t)32 * VT * 512;           // 606208
    const size_t W2T_ALLOC  = W2T_ELEMS + (size_t)VT * 512;    // +1 chunk slack
    const size_t XA_ELEMS   = (size_t)SEQ * MT * 2 * 512;      // 15466496
    const size_t HA_ELEMS   = (size_t)MT * 32 * 512;           // 4194304
    const size_t HA_ALLOC   = HA_ELEMS + 512;                  // +1 tile slack

    unsigned short* wk  = (unsigned short*)take(WK_ELEMS * 2);
    unsigned short* wr  = (unsigned short*)take(WR_ALLOC * 2);
    unsigned short* w2t = (unsigned short*)take(W2T_ALLOC * 2);
    unsigned short* xa  = (unsigned short*)take(XA_ELEMS * 2);
    float*          c   = (float*)take((size_t)BATCH * UNITS * 4);
    unsigned short* ha0 = (unsigned short*)take(HA_ALLOC * 2);
    unsigned short* ha1 = (unsigned short*)take(HA_ALLOC * 2);

    hipMemsetAsync(c,   0, (size_t)BATCH * UNITS * 4, stream);
    hipMemsetAsync(ha0, 0, HA_ELEMS * 2, stream);

    conv_w_tiles<<<(unsigned)((WK_ELEMS + 255) / 256), 256, 0, stream>>>(
        kernel, FEAT, GATES4U, GATES4U, 256, (int)WK_ELEMS, wk);
    conv_w_tiles<<<(unsigned)((WR_ELEMS + 255) / 256), 256, 0, stream>>>(
        rec, UNITS, GATES4U, GATES4U, 256, (int)WR_ELEMS, wr);
    conv_w_tiles<<<(unsigned)((W2T_ELEMS + 255) / 256), 256, 0, stream>>>(
        w2, UNITS, VOCAB, VOCAB, VT, (int)W2T_ELEMS, w2t);
    conv_x_tiles<<<(unsigned)((XA_ELEMS + 255) / 256), 256, 0, stream>>>(
        x, xa, (int)XA_ELEMS);

    dim3 sgrid(BATCH / 64, UNITS / 32);
    for (int t = 0; t < SEQ; ++t) {
        unsigned short* hp = (t & 1) ? ha1 : ha0;
        unsigned short* hn = (t & 1) ? ha0 : ha1;
        lstm_step_kernel<<<sgrid, 256, 0, stream>>>(
            xa + (size_t)t * MT * 2 * 512, wk, hp, wr, bias, c, hn);
    }
    unsigned short* hlast = (SEQ & 1) ? ha1 : ha0;  // SEQ=59 -> ha1
    dense_softmax_kernel<<<MT, 256, 0, stream>>>(hlast, w2t, b2, out);
}
